// SwinGraphBlock_74534862455396
// MI455X (gfx1250) — matrix operand order repaired and checked by reading, not compiled, not run on hardware
//
#include <hip/hip_runtime.h>
#include <math.h>
#include <stdint.h>

typedef __bf16 bf16;
typedef __attribute__((ext_vector_type(16))) __bf16 v16bf;
typedef __attribute__((ext_vector_type(8)))  float  v8f;

union AB { v16bf v; bf16 h[16]; uint32_t u[8]; };
union F8 { v8f v; float f[8]; };

#define N_NODES 2048
#define C_DIM   256
#define K_CL    16
#define S_SZ    128
#define H_HEADS 8
#define DH      64
#define INNER   512
#define SCALE_ATTN 0.125f
#define VT_LD   130   // padded LDS row stride for transposed v (bank-conflict free, 4B aligned)

// ---------------------------------------------------------------------------
// Small prep kernels
// ---------------------------------------------------------------------------
__global__ void wconv_kernel(const float* __restrict__ W, bf16* __restrict__ WT,
                             int Cin, int Cout) {
  int idx = blockIdx.x * 256 + threadIdx.x;
  if (idx >= Cin * Cout) return;
  int c = idx / Cout, o = idx - c * Cout;
  WT[(size_t)o * Cin + c] = (bf16)W[idx];           // store W^T in bf16
}

__global__ void zero_u32_kernel(uint32_t* p, int n) {
  int i = blockIdx.x * 256 + threadIdx.x;
  if (i < n) p[i] = 0u;
}

__global__ void e_bias_kernel(const float* __restrict__ adj_emb,
                              const float* __restrict__ We,
                              const float* __restrict__ be,
                              float* __restrict__ e01) {
  int a = blockIdx.x;            // 0 or 1
  int j = threadIdx.x;           // 0..511
  float s = 0.f;
  for (int c = 0; c < C_DIM; ++c) s += adj_emb[a * C_DIM + c] * We[(size_t)c * INNER + j];
  e01[a * INNER + j] = s + be[j];
}

__global__ void maps_kernel(const int* __restrict__ clusters,
                            int* __restrict__ node2c, int* __restrict__ node2p) {
  int idx = blockIdx.x * 256 + threadIdx.x;      // 0..2047 = k*S+s
  int n = clusters[idx];
  node2c[n] = idx >> 7;
  node2p[n] = idx & (S_SZ - 1);
}

__global__ void adj_kernel(const int* __restrict__ ei,
                           const int* __restrict__ node2c,
                           const int* __restrict__ node2p,
                           uint32_t* __restrict__ adjbits, int E) {
  int e = blockIdx.x * 256 + threadIdx.x;
  if (e >= E) return;
  int u = ei[e], v = ei[E + e];
  int cu = node2c[u];
  if (cu == node2c[v]) {
    int pu = node2p[u], pv = node2p[v];
    atomicOr(&adjbits[((size_t)cu * S_SZ + pu) * 4 + (pv >> 5)], 1u << (pv & 31));
  }
}

// ---------------------------------------------------------------------------
// LayerNorm (+ optional gather via clusters, optional f32 passthrough)
// ---------------------------------------------------------------------------
__global__ __launch_bounds__(256) void ln_kernel(const float* __restrict__ x,
                                                 const int* __restrict__ clusters,
                                                 const float* __restrict__ g,
                                                 const float* __restrict__ b,
                                                 float* __restrict__ nodes_out,
                                                 bf16* __restrict__ h_out) {
  int r = blockIdx.x, c = threadIdx.x;
  int src = clusters ? clusters[r] : r;
  float xv = x[(size_t)src * C_DIM + c];
  __shared__ float s1[256], s2[256];
  s1[c] = xv; s2[c] = xv * xv;
  __syncthreads();
  for (int off = 128; off > 0; off >>= 1) {
    if (c < off) { s1[c] += s1[c + off]; s2[c] += s2[c + off]; }
    __syncthreads();
  }
  float mean = s1[0] * (1.f / C_DIM);
  float var  = s2[0] * (1.f / C_DIM) - mean * mean;
  float rstd = rsqrtf(var + 1e-5f);
  if (nodes_out) nodes_out[(size_t)r * C_DIM + c] = xv;
  h_out[(size_t)r * C_DIM + c] = (bf16)((xv - mean) * rstd * g[c] + b[c]);
}

// ---------------------------------------------------------------------------
// bf16 WMMA GEMM:  out[M,N] = A[M,K] @ BT[N,K]^T + bias
// mode 0: f32 out | mode 1: bf16 out | mode 2: exact GELU then bf16 out
// grid = (M/16, N/256), block = 128 (4 waves, each wave 4 N-tiles of 16)
// ---------------------------------------------------------------------------
__global__ __launch_bounds__(128) void wmma_gemm_kernel(
    const bf16* __restrict__ A, const bf16* __restrict__ BT,
    const float* __restrict__ bias, float* __restrict__ outF,
    bf16* __restrict__ outB, int M, int Kdim, int N, int mode) {
  const int lane = threadIdx.x & 31, wave = threadIdx.x >> 5;
  const int half = lane >> 4, lm = lane & 15;
  const int m0 = blockIdx.x * 16;
  const int n0 = blockIdx.y * 256 + wave * 64;

  F8 acc[4];
#pragma unroll
  for (int t = 0; t < 4; ++t)
#pragma unroll
    for (int i = 0; i < 8; ++i) acc[t].f[i] = 0.f;

  for (int kk = 0; kk < Kdim; kk += 32) {
    AB a;
    const bf16* arow = A + (size_t)(m0 + lm) * Kdim + kk;
#pragma unroll
    for (int r = 0; r < 8; ++r) {
      int Ka = (r >> 2) * 16 + half * 8 + (r & 3) * 2;
      a.u[r] = *reinterpret_cast<const uint32_t*>(arow + Ka);
    }
#pragma unroll
    for (int t = 0; t < 4; ++t) {
      AB b;
      const bf16* brow = BT + (size_t)(n0 + t * 16 + lm) * Kdim + kk;
#pragma unroll
      for (int r = 0; r < 8; ++r)
        b.u[r] = *reinterpret_cast<const uint32_t*>(brow + (r >> 2) * 16 + half * 8 + (r & 3) * 2);
      acc[t].v = __builtin_amdgcn_wmma_f32_16x16x32_bf16(
          false, a.v, false, b.v, (short)0, acc[t].v, false, false);
    }
  }
#pragma unroll
  for (int t = 0; t < 4; ++t)
#pragma unroll
    for (int r = 0; r < 8; ++r) {
      int row = m0 + r + 8 * half;
      int col = n0 + t * 16 + lm;
      float v = acc[t].f[r] + bias[col];
      if (mode == 2) v = 0.5f * v * (1.f + erff(v * 0.70710678118654752f));
      if (mode == 0) outF[(size_t)row * N + col] = v;
      else           outB[(size_t)row * N + col] = (bf16)v;
    }
}

// ---------------------------------------------------------------------------
// Rotary (interleaved pairs) applied in place to q and the k half of kv
// ---------------------------------------------------------------------------
__global__ __launch_bounds__(256) void rotary_kernel(bf16* __restrict__ q,
                                                     bf16* __restrict__ kv) {
  int row = blockIdx.x;
  int pp = threadIdx.x;                 // 256 pairs = inner/2
  int h = pp >> 5, p = pp & 31;
  int s = row & (S_SZ - 1);
  float inv = powf(10000.f, -(float)p * (1.f / 32.f));
  float ang = (float)s * inv;
  float cs, sn; __sincosf(ang, &sn, &cs);
  size_t bq = (size_t)row * INNER + h * DH + 2 * p;
  float x1 = (float)q[bq], x2 = (float)q[bq + 1];
  q[bq]     = (bf16)(x1 * cs - x2 * sn);
  q[bq + 1] = (bf16)(x2 * cs + x1 * sn);
  size_t bk = (size_t)row * (2 * INNER) + h * DH + 2 * p;  // k half of kv
  x1 = (float)kv[bk]; x2 = (float)kv[bk + 1];
  kv[bk]     = (bf16)(x1 * cs - x2 * sn);
  kv[bk + 1] = (bf16)(x2 * cs + x1 * sn);
}

// qe[row,h,{0,1}] = q_rot[row,h,:] . e{0,1}[h,:]
__global__ __launch_bounds__(256) void qe_kernel(const bf16* __restrict__ q,
                                                 const float* __restrict__ e01,
                                                 float* __restrict__ qe) {
  int row = blockIdx.x;
  int lane = threadIdx.x & 31, h = threadIdx.x >> 5;
  size_t base = (size_t)row * INNER + h * DH;
  float q0 = (float)q[base + lane], q1 = (float)q[base + lane + 32];
  float a0 = q0 * e01[h * DH + lane] + q1 * e01[h * DH + lane + 32];
  float a1 = q0 * e01[INNER + h * DH + lane] + q1 * e01[INNER + h * DH + lane + 32];
  for (int off = 16; off > 0; off >>= 1) {
    a0 += __shfl_xor(a0, off, 32);
    a1 += __shfl_xor(a1, off, 32);
  }
  if (lane == 0) {
    qe[((size_t)row * H_HEADS + h) * 2]     = a0;
    qe[((size_t)row * H_HEADS + h) * 2 + 1] = a1;
  }
}

// ---------------------------------------------------------------------------
// Fused attention per (cluster, head): 128x128 logits, register softmax,
// adjacency bias, attn@v + edge term. 8 waves, 16 rows each, 32 WMMA/wave.
// v is staged TRANSPOSED in LDS (padded stride) so attn@v B-fragments are
// contiguous 4B words, bank-conflict free.
// ---------------------------------------------------------------------------
__global__ __launch_bounds__(256) void attn_kernel(
    const bf16* __restrict__ qb, const bf16* __restrict__ kvb,
    const float* __restrict__ qe, const uint32_t* __restrict__ adjbits,
    const float* __restrict__ e01, bf16* __restrict__ outb) {
  __shared__ __align__(16) bf16 s_attn[8 * 16 * S_SZ];   // 32 KB, wave-private slabs
  __shared__ __align__(16) bf16 s_vt[DH * VT_LD];        // v transposed, ~16.3 KB
  __shared__ uint32_t s_adj[S_SZ * 4];                   // 2 KB

  const int kcl = blockIdx.x, h = blockIdx.y;
  const int tid = threadIdx.x;
  const int lane = tid & 31, wave = tid >> 5;
  const int half = lane >> 4, lm = lane & 15;
  const int rowbase = kcl * S_SZ;

  // stage v transposed: s_vt[d * VT_LD + j] = v[j, d]
  // consecutive tid -> consecutive d: global reads coalesced, LDS writes hit
  // distinct banks (row stride 260 B = 65 words).
  for (int idx = tid; idx < S_SZ * DH; idx += 256) {
    int j = idx >> 6, d = idx & 63;
    s_vt[d * VT_LD + j] = kvb[(size_t)(rowbase + j) * (2 * INNER) + INNER + h * DH + d];
  }
  for (int idx = tid; idx < S_SZ * 4; idx += 256)
    s_adj[idx] = adjbits[(size_t)kcl * S_SZ * 4 + idx];
  __syncthreads();

  const int i0 = wave * 16;

  // ---- sim = q @ k^T over this wave's 16 rows x 128 cols ----
  F8 accS[8];
#pragma unroll
  for (int t = 0; t < 8; ++t)
#pragma unroll
    for (int i = 0; i < 8; ++i) accS[t].f[i] = 0.f;

  for (int kk = 0; kk < DH; kk += 32) {
    AB a;
    const bf16* arow = qb + (size_t)(rowbase + i0 + lm) * INNER + h * DH + kk;
#pragma unroll
    for (int r = 0; r < 8; ++r) {
      int Ka = (r >> 2) * 16 + half * 8 + (r & 3) * 2;
      a.u[r] = *reinterpret_cast<const uint32_t*>(arow + Ka);
    }
#pragma unroll
    for (int t = 0; t < 8; ++t) {
      AB b;   // B[K=d, N=j] = k[j0+N, kk+K]; pairs contiguous along d
      const bf16* brow = kvb + (size_t)(rowbase + t * 16 + lm) * (2 * INNER) + h * DH + kk;
#pragma unroll
      for (int r = 0; r < 8; ++r)
        b.u[r] = *reinterpret_cast<const uint32_t*>(brow + (r >> 2) * 16 + half * 8 + (r & 3) * 2);
      accS[t].v = __builtin_amdgcn_wmma_f32_16x16x32_bf16(
          false, a.v, false, b.v, (short)0, accS[t].v, false, false);
    }
  }

  // ---- adjacency bias + scale + softmax (in C-fragment layout) ----
  float qe0r[8], qe1r[8], mx[8], sum[8], wsm[8];
#pragma unroll
  for (int r = 0; r < 8; ++r) {
    int i = i0 + r + 8 * half;
    qe0r[r] = qe[((size_t)(rowbase + i) * H_HEADS + h) * 2];
    qe1r[r] = qe[((size_t)(rowbase + i) * H_HEADS + h) * 2 + 1];
    mx[r] = -1e30f; sum[r] = 0.f; wsm[r] = 0.f;
  }
#pragma unroll
  for (int t = 0; t < 8; ++t)
#pragma unroll
    for (int r = 0; r < 8; ++r) {
      int i = i0 + r + 8 * half, j = t * 16 + lm;
      uint32_t bit = (s_adj[i * 4 + (j >> 5)] >> (j & 31)) & 1u;
      float sv = (accS[t].f[r] + (bit ? qe1r[r] : qe0r[r])) * SCALE_ATTN;
      accS[t].f[r] = sv;
      mx[r] = fmaxf(mx[r], sv);
    }
#pragma unroll
  for (int r = 0; r < 8; ++r)
    for (int off = 8; off > 0; off >>= 1) mx[r] = fmaxf(mx[r], __shfl_xor(mx[r], off, 32));
#pragma unroll
  for (int t = 0; t < 8; ++t)
#pragma unroll
    for (int r = 0; r < 8; ++r) {
      int i = i0 + r + 8 * half, j = t * 16 + lm;
      uint32_t bit = (s_adj[i * 4 + (j >> 5)] >> (j & 31)) & 1u;
      float p = __expf(accS[t].f[r] - mx[r]);
      accS[t].f[r] = p;
      sum[r] += p;
      if (bit) wsm[r] += p;
    }
#pragma unroll
  for (int r = 0; r < 8; ++r)
    for (int off = 8; off > 0; off >>= 1) {
      sum[r] += __shfl_xor(sum[r], off, 32);
      wsm[r] += __shfl_xor(wsm[r], off, 32);
    }
  float wadj[8], rs[8];
#pragma unroll
  for (int r = 0; r < 8; ++r) { rs[r] = 1.f / sum[r]; wadj[r] = wsm[r] * rs[r]; }

  // ---- restripe normalized attn (C layout -> A layout) via wave-private LDS ----
  bf16* sa = s_attn + wave * 16 * S_SZ;
#pragma unroll
  for (int t = 0; t < 8; ++t)
#pragma unroll
    for (int r = 0; r < 8; ++r) {
      int il = r + 8 * half, j = t * 16 + lm;
      sa[il * S_SZ + j] = (bf16)(accS[t].f[r] * rs[r]);
    }

  // ---- out = attn @ v ----
  F8 accO[4];
#pragma unroll
  for (int t = 0; t < 4; ++t)
#pragma unroll
    for (int i = 0; i < 8; ++i) accO[t].f[i] = 0.f;

  for (int kk = 0; kk < S_SZ; kk += 32) {
    AB a;
    const bf16* arow = sa + lm * S_SZ + kk;
#pragma unroll
    for (int r = 0; r < 8; ++r) {
      int Ka = (r >> 2) * 16 + half * 8 + (r & 3) * 2;
      a.u[r] = *reinterpret_cast<const uint32_t*>(arow + Ka);
    }
#pragma unroll
    for (int t2 = 0; t2 < 4; ++t2) {
      AB b;   // B[K=j, N=d] = s_vt[d * VT_LD + (kk+K)]; K-pairs contiguous
      const bf16* vtrow = s_vt + (t2 * 16 + lm) * VT_LD + kk;
#pragma unroll
      for (int r = 0; r < 8; ++r)
        b.u[r] = *reinterpret_cast<const uint32_t*>(vtrow + (r >> 2) * 16 + half * 8 + (r & 3) * 2);
      accO[t2].v = __builtin_amdgcn_wmma_f32_16x16x32_bf16(
          false, a.v, false, b.v, (short)0, accO[t2].v, false, false);
    }
  }

  // ---- add attn@e term and write bf16 ----
#pragma unroll
  for (int t2 = 0; t2 < 4; ++t2)
#pragma unroll
    for (int r = 0; r < 8; ++r) {
      int i = i0 + r + 8 * half, d = t2 * 16 + lm;
      float ev0 = e01[h * DH + d], ev1 = e01[INNER + h * DH + d];
      float v = accO[t2].f[r] + ev0 * (1.f - wadj[r]) + ev1 * wadj[r];
      outb[(size_t)(rowbase + i) * INNER + h * DH + d] = (bf16)v;
    }
}

// ---------------------------------------------------------------------------
// Gated residual: gate = sigmoid([x, res, x-res] @ w);  out = x*g + res*(1-g)
// ---------------------------------------------------------------------------
__global__ __launch_bounds__(256) void gate_kernel(const float* __restrict__ xin,
                                                   const float* __restrict__ res,
                                                   const float* __restrict__ gw,
                                                   float* __restrict__ out) {
  int r = blockIdx.x, c = threadIdx.x;
  float a = xin[(size_t)r * C_DIM + c], b = res[(size_t)r * C_DIM + c];
  __shared__ float s1[256];
  s1[c] = a * gw[c] + b * gw[C_DIM + c] + (a - b) * gw[2 * C_DIM + c];
  __syncthreads();
  for (int off = 128; off > 0; off >>= 1) {
    if (c < off) s1[c] += s1[c + off];
    __syncthreads();
  }
  float gate = 1.f / (1.f + __expf(-s1[0]));
  out[(size_t)r * C_DIM + c] = a * gate + b * (1.f - gate);
}

__global__ void copy_kernel(const float* __restrict__ src, float* __restrict__ dst, int n) {
  int i = blockIdx.x * 256 + threadIdx.x;
  if (i < n) dst[i] = src[i];
}

__global__ __launch_bounds__(256) void scatter_kernel(const int* __restrict__ clusters,
                                                      const float* __restrict__ nodes2,
                                                      float* __restrict__ out) {
  int r = blockIdx.x, c = threadIdx.x;
  int n = clusters[r];
  out[(size_t)n * C_DIM + c] += nodes2[(size_t)r * C_DIM + c];
}

// ---------------------------------------------------------------------------
extern "C" void kernel_launch(void* const* d_in, const int* in_sizes, int n_in,
                              void* d_out, int out_size, void* d_ws, size_t ws_size,
                              hipStream_t stream) {
  const float* x        = (const float*)d_in[0];
  const int*   edge_idx = (const int*)  d_in[1];
  const int*   clusters = (const int*)  d_in[2];
  const float* adj_emb  = (const float*)d_in[3];
  const float* ln1_g    = (const float*)d_in[4];
  const float* ln1_b    = (const float*)d_in[5];
  const float* Wq       = (const float*)d_in[6];
  const float* bq       = (const float*)d_in[7];
  const float* Wkv      = (const float*)d_in[8];
  const float* bkv      = (const float*)d_in[9];
  const float* We       = (const float*)d_in[10];
  const float* be       = (const float*)d_in[11];
  const float* Wo       = (const float*)d_in[12];
  const float* bo       = (const float*)d_in[13];
  const float* g1w      = (const float*)d_in[14];
  const float* ln2_g    = (const float*)d_in[15];
  const float* ln2_b    = (const float*)d_in[16];
  const float* W1       = (const float*)d_in[17];
  const float* b1       = (const float*)d_in[18];
  const float* W2       = (const float*)d_in[19];
  const float* b2       = (const float*)d_in[20];
  const float* g2w      = (const float*)d_in[21];
  float* out = (float*)d_out;
  const int E = in_sizes[1] / 2;

  char* ws = (char*)d_ws;
  size_t off = 0;
  auto alloc = [&](size_t bytes) -> char* {
    char* p = ws + off;
    off += (bytes + 255) & ~(size_t)255;
    return p;
  };
  int*      node2c   = (int*)alloc(N_NODES * 4);
  int*      node2p   = (int*)alloc(N_NODES * 4);
  uint32_t* adjbits  = (uint32_t*)alloc(K_CL * S_SZ * 4 * 4);
  float*    nodes    = (float*)alloc((size_t)N_NODES * C_DIM * 4);
  bf16*     h_bf     = (bf16*)alloc((size_t)N_NODES * C_DIM * 2);
  bf16*     WqT      = (bf16*)alloc((size_t)INNER * C_DIM * 2);
  bf16*     WkvT     = (bf16*)alloc((size_t)2 * INNER * C_DIM * 2);
  bf16*     WoT      = (bf16*)alloc((size_t)C_DIM * INNER * 2);
  bf16*     W1T      = (bf16*)alloc((size_t)4 * C_DIM * C_DIM * 2);
  bf16*     W2T      = (bf16*)alloc((size_t)C_DIM * 4 * C_DIM * 2);
  bf16*     q_bf     = (bf16*)alloc((size_t)N_NODES * INNER * 2);
  bf16*     kv_bf    = (bf16*)alloc((size_t)N_NODES * 2 * INNER * 2);
  float*    e01      = (float*)alloc(2 * INNER * 4);
  float*    qe       = (float*)alloc((size_t)N_NODES * H_HEADS * 2 * 4);
  bf16*     ao_bf    = (bf16*)alloc((size_t)N_NODES * INNER * 2);
  float*    o_f      = (float*)alloc((size_t)N_NODES * C_DIM * 4);
  float*    nodes1   = (float*)alloc((size_t)N_NODES * C_DIM * 4);
  bf16*     h2_bf    = (bf16*)alloc((size_t)N_NODES * C_DIM * 2);
  bf16*     ff1_bf   = (bf16*)alloc((size_t)N_NODES * 4 * C_DIM * 2);
  float*    ff2_f    = (float*)alloc((size_t)N_NODES * C_DIM * 4);
  float*    nodes2   = (float*)alloc((size_t)N_NODES * C_DIM * 4);

  // --- weight prep (bf16 transposed) + edge-bias vectors ---
  wconv_kernel<<<(C_DIM * INNER + 255) / 256, 256, 0, stream>>>(Wq, WqT, C_DIM, INNER);
  wconv_kernel<<<(C_DIM * 2 * INNER + 255) / 256, 256, 0, stream>>>(Wkv, WkvT, C_DIM, 2 * INNER);
  wconv_kernel<<<(INNER * C_DIM + 255) / 256, 256, 0, stream>>>(Wo, WoT, INNER, C_DIM);
  wconv_kernel<<<(C_DIM * 4 * C_DIM + 255) / 256, 256, 0, stream>>>(W1, W1T, C_DIM, 4 * C_DIM);
  wconv_kernel<<<(4 * C_DIM * C_DIM + 255) / 256, 256, 0, stream>>>(W2, W2T, 4 * C_DIM, C_DIM);
  e_bias_kernel<<<2, 512, 0, stream>>>(adj_emb, We, be, e01);

  // --- graph structure ---
  maps_kernel<<<N_NODES / 256, 256, 0, stream>>>(clusters, node2c, node2p);
  zero_u32_kernel<<<(K_CL * S_SZ * 4 + 255) / 256, 256, 0, stream>>>(adjbits, K_CL * S_SZ * 4);
  adj_kernel<<<(E + 255) / 256, 256, 0, stream>>>(edge_idx, node2c, node2p, adjbits, E);

  // --- attention branch ---
  ln_kernel<<<N_NODES, 256, 0, stream>>>(x, clusters, ln1_g, ln1_b, nodes, h_bf);
  wmma_gemm_kernel<<<dim3(N_NODES / 16, INNER / 256), 128, 0, stream>>>(
      h_bf, WqT, bq, nullptr, q_bf, N_NODES, C_DIM, INNER, 1);
  wmma_gemm_kernel<<<dim3(N_NODES / 16, 2 * INNER / 256), 128, 0, stream>>>(
      h_bf, WkvT, bkv, nullptr, kv_bf, N_NODES, C_DIM, 2 * INNER, 1);
  rotary_kernel<<<N_NODES, 256, 0, stream>>>(q_bf, kv_bf);
  qe_kernel<<<N_NODES, 256, 0, stream>>>(q_bf, e01, qe);
  attn_kernel<<<dim3(K_CL, H_HEADS), 256, 0, stream>>>(q_bf, kv_bf, qe, adjbits, e01, ao_bf);
  wmma_gemm_kernel<<<dim3(N_NODES / 16, 1), 128, 0, stream>>>(
      ao_bf, WoT, bo, o_f, nullptr, N_NODES, INNER, C_DIM, 0);
  gate_kernel<<<N_NODES, 256, 0, stream>>>(o_f, nodes, g1w, nodes1);

  // --- feedforward branch ---
  ln_kernel<<<N_NODES, 256, 0, stream>>>(nodes1, nullptr, ln2_g, ln2_b, nullptr, h2_bf);
  wmma_gemm_kernel<<<dim3(N_NODES / 16, 4 * C_DIM / 256), 128, 0, stream>>>(
      h2_bf, W1T, b1, nullptr, ff1_bf, N_NODES, C_DIM, 4 * C_DIM, 2);
  wmma_gemm_kernel<<<dim3(N_NODES / 16, 1), 128, 0, stream>>>(
      ff1_bf, W2T, b2, ff2_f, nullptr, N_NODES, 4 * C_DIM, C_DIM, 0);
  gate_kernel<<<N_NODES, 256, 0, stream>>>(ff2_f, nodes1, g2w, nodes2);

  // --- residual scatter back ---
  copy_kernel<<<(N_NODES * C_DIM + 255) / 256, 256, 0, stream>>>(x, out, N_NODES * C_DIM);
  scatter_kernel<<<N_NODES, 256, 0, stream>>>(clusters, nodes2, out);
}
